// MyTransformerEncoderLayer_70626442215439
// MI455X (gfx1250) — matrix operand order repaired and checked by reading, not compiled, not run on hardware
//
#include <hip/hip_runtime.h>
#include <hip/hip_bf16.h>
#include <math.h>

#define D_MODEL 512
#define SEQLEN  1024
#define BATCH   8
#define NHEADS  8
#define DHEAD   64
#define DFF     2048
#define NTOK    (BATCH * SEQLEN)   // 8192 tokens

typedef __attribute__((ext_vector_type(16))) __bf16 v16bf;
typedef __attribute__((ext_vector_type(8)))  float  v8f;

union FragBF { v16bf v; unsigned u[8]; };

__device__ __forceinline__ unsigned short f2bf(float f) {
  union { float f; unsigned u; } x; x.f = f;
  unsigned r = x.u + 0x7fffu + ((x.u >> 16) & 1u);   // round-to-nearest-even
  return (unsigned short)(r >> 16);
}
__device__ __forceinline__ unsigned pack2bf(float a, float b) {
  return (unsigned)f2bf(a) | ((unsigned)f2bf(b) << 16);
}

// ---- DPP16 butterfly reductions over each 16-lane half (no LDS, no waits) ----
template <int CTRL>
__device__ __forceinline__ float dpp_mv(float x) {
  int i = __builtin_bit_cast(int, x);
  int r = __builtin_amdgcn_update_dpp(i, i, CTRL, 0xf, 0xf, false);
  return __builtin_bit_cast(float, r);
}
__device__ __forceinline__ float rowmax16(float t) {
  t = fmaxf(t, dpp_mv<0xB1>(t));    // quad_perm xor1
  t = fmaxf(t, dpp_mv<0x4E>(t));    // quad_perm xor2
  t = fmaxf(t, dpp_mv<0x141>(t));   // row_half_mirror (xor7 in 8)
  t = fmaxf(t, dpp_mv<0x140>(t));   // row_mirror (xor15 in 16)
  return t;
}
__device__ __forceinline__ float rowsum16(float t) {
  t += dpp_mv<0xB1>(t);
  t += dpp_mv<0x4E>(t);
  t += dpp_mv<0x141>(t);
  t += dpp_mv<0x140>(t);
  return t;
}

// ---------------- elementwise f32 -> bf16 ----------------
__global__ void cvt_bf16_kernel(const float* __restrict__ in,
                                unsigned short* __restrict__ out, int n) {
  int i = blockIdx.x * blockDim.x + threadIdx.x;
  if (i < n) out[i] = f2bf(in[i]);
}

// --- pack W[K][N] f32 -> fragment-linear bf16 pairs:
//     Bq[((k/32)*N + n)*16 + half*8 + p] = {W[k0+2p+16h][n], W[k0+2p+1+16h][n]}
__global__ void pack_weight_kernel(const float* __restrict__ W,
                                   unsigned* __restrict__ Bq, int K, int N) {
  int i = blockIdx.x * blockDim.x + threadIdx.x;
  int total = (K >> 5) * N * 16;
  if (i >= total) return;
  int p    = i & 7;
  int half = (i >> 3) & 1;
  int n    = (i >> 4) % N;
  int kb   = (i >> 4) / N;
  int k    = kb * 32 + 2 * p + 16 * half;
  Bq[i] = pack2bf(W[(size_t)k * N + n], W[(size_t)(k + 1) * N + n]);
}

// ---------------- bf16 WMMA GEMM: wave computes 32 rows x 64 cols ----------------
// EPI 0: bf16 row-major out   EPI 1: leaky_relu -> bf16   EPI 2: f32 = resid + val
// EPI 3: fragment-linear key-pair-packed out (for attention V)
template <int EPI>
__global__ void gemm_wmma_kernel(const unsigned short* __restrict__ A,
                                 const unsigned* __restrict__ Bq,
                                 const float* __restrict__ bias,
                                 int M, int N, int K,
                                 unsigned short* __restrict__ Ob,
                                 float* __restrict__ Of,
                                 const float* __restrict__ resid) {
  const int lane = threadIdx.x & 31;
  const int wave = threadIdx.x >> 5;
  const int row0 = blockIdx.y * 128 + wave * 32;
  const int col0 = blockIdx.x * 64;
  const int m    = lane & 15;
  const int half = lane >> 4;

  v8f acc[2][4] = {};

  const unsigned short* arow0 = A + (size_t)(row0 + m) * K;
  const unsigned short* arow1 = A + (size_t)(row0 + 16 + m) * K;
  for (int kb = 0; kb < K; kb += 32) {
    FragBF a[2];
#pragma unroll
    for (int p = 0; p < 8; ++p) {
      int kl = 2 * p + 16 * half;   // A 16x32 layout
      a[0].u[p] = *(const unsigned*)(arow0 + kb + kl);
      a[1].u[p] = *(const unsigned*)(arow1 + kb + kl);
    }
#pragma unroll
    for (int t = 0; t < 4; ++t) {
      const unsigned* bptr =
          Bq + ((size_t)(kb >> 5) * N + col0 + t * 16 + m) * 16 + half * 8;
      FragBF b;
#pragma unroll
      for (int p = 0; p < 8; ++p) b.u[p] = bptr[p];   // 2 x b128
      acc[0][t] = __builtin_amdgcn_wmma_f32_16x16x32_bf16(
          false, a[0].v, false, b.v, (short)0, acc[0][t], false, false);
      acc[1][t] = __builtin_amdgcn_wmma_f32_16x16x32_bf16(
          false, a[1].v, false, b.v, (short)0, acc[1][t], false, false);
    }
  }

  unsigned* Ov = (unsigned*)Ob;
#pragma unroll
  for (int i = 0; i < 2; ++i) {
#pragma unroll
    for (int t = 0; t < 4; ++t) {
      int col = col0 + t * 16 + m;
      if (EPI == 3) {
#pragma unroll
        for (int r = 0; r < 8; r += 2) {
          int tok = row0 + i * 16 + 8 * half + r;     // rows tok, tok+1 in this lane
          float v0 = acc[i][t][r] + bias[col];
          float v1 = acc[i][t][r + 1] + bias[col];
          size_t idx = ((size_t)(tok >> 5) * N + col) * 16 +
                       ((tok >> 4) & 1) * 8 + ((tok & 15) >> 1);
          Ov[idx] = pack2bf(v0, v1);
        }
      } else {
#pragma unroll
        for (int r = 0; r < 8; ++r) {
          int row = row0 + i * 16 + r + 8 * half;
          float v = acc[i][t][r] + bias[col];
          if (EPI == 1) v = (v > 0.0f) ? v : 0.01f * v;
          if (EPI == 2) Of[(size_t)row * N + col] = resid[(size_t)row * N + col] + v;
          else          Ob[(size_t)row * N + col] = f2bf(v);
        }
      }
    }
  }
}

// ---------------- flash attention, one wave per (b,h,16-query tile) ----------------
__global__ void __launch_bounds__(32, 1)
attn_flash_kernel(const unsigned short* __restrict__ Qb,
                  const unsigned short* __restrict__ Kb,
                  const unsigned* __restrict__ Vq,
                  float* __restrict__ attn_out) {
  const int lane = threadIdx.x;        // blockDim.x == 32
  const int qt   = blockIdx.x;         // query tile (S/16)
  const int bh   = blockIdx.y;         // b*NHEADS + h
  const int b    = bh >> 3;
  const int h    = bh & 7;
  const int n    = lane & 15;
  const int half = lane >> 4;
  const int D = D_MODEL, S = SEQLEN;

  // Q tile [16 x 64] as two A-fragments (dh 0..31, 32..63)
  FragBF aq[2];
  const unsigned short* qrow = Qb + (size_t)(b * S + qt * 16 + n) * D + h * DHEAD;
#pragma unroll
  for (int d = 0; d < 2; ++d) {
#pragma unroll
    for (int p = 0; p < 8; ++p) {
      int kl = 2 * p + 16 * half;
      aq[d].u[p] = *(const unsigned*)(qrow + d * 32 + kl);
    }
  }

  v8f o[4] = {};
  float mrow[8], lrow[8];
#pragma unroll
  for (int r = 0; r < 8; ++r) { mrow[r] = -1e30f; lrow[r] = 0.0f; }

  __shared__ unsigned short pbuf16[16 * 32];   // P tile, bf16

  for (int k0 = 0; k0 < S; k0 += 32) {
    // scores S[16 x 32] = Q x K^T (two 16x16 D tiles)
    v8f s[2] = {};
#pragma unroll
    for (int j = 0; j < 2; ++j) {
      const unsigned short* krow =
          Kb + (size_t)(b * S + k0 + j * 16 + n) * D + h * DHEAD;
#pragma unroll
      for (int d = 0; d < 2; ++d) {
        FragBF bk;
#pragma unroll
        for (int p = 0; p < 8; ++p) {
          int kk = 2 * p + 16 * half;              // dh index within 32-chunk
          bk.u[p] = *(const unsigned*)(krow + d * 32 + kk);
        }
        s[j] = __builtin_amdgcn_wmma_f32_16x16x32_bf16(
            false, aq[d].v, false, bk.v, (short)0, s[j], false, false);
      }
    }

    // streaming softmax update (rows live across 16-lane halves, DPP reductions)
    float alpha[8];
#pragma unroll
    for (int r = 0; r < 8; ++r) {
      float t = rowmax16(fmaxf(s[0][r], s[1][r]));
      float mnew = fmaxf(mrow[r], t);
      alpha[r] = __expf(mrow[r] - mnew);
      float p0 = __expf(s[0][r] - mnew);
      float p1 = __expf(s[1][r] - mnew);
      s[0][r] = p0; s[1][r] = p1;
      lrow[r] = lrow[r] * alpha[r] + rowsum16(p0 + p1);
      mrow[r] = mnew;
    }
#pragma unroll
    for (int t4 = 0; t4 < 4; ++t4)
#pragma unroll
      for (int r = 0; r < 8; ++r) o[t4][r] *= alpha[r];

    // P (D-tile layout) -> LDS (bf16) -> A-fragment layout (dword pair reads)
#pragma unroll
    for (int r = 0; r < 8; ++r) {
      int row = r + 8 * half;
      pbuf16[row * 32 + n]      = f2bf(s[0][r]);
      pbuf16[row * 32 + 16 + n] = f2bf(s[1][r]);
    }
    __syncthreads();
    FragBF pa;
#pragma unroll
    for (int p = 0; p < 8; ++p) {
      int kl = 2 * p + 16 * half;
      pa.u[p] = *(const unsigned*)&pbuf16[n * 32 + kl];
    }
    __syncthreads();

    // O += P x V ; V fragments are fragment-linear packed: 2 x b128 per fragment
    const unsigned* vbase =
        Vq + ((size_t)(b * 32 + (k0 >> 5)) * D + h * DHEAD) * 16 + half * 8;
#pragma unroll
    for (int t4 = 0; t4 < 4; ++t4) {
      const unsigned* vptr = vbase + (size_t)(t4 * 16 + n) * 16;
      FragBF bv;
#pragma unroll
      for (int p = 0; p < 8; ++p) bv.u[p] = vptr[p];
      o[t4] = __builtin_amdgcn_wmma_f32_16x16x32_bf16(
          false, pa.v, false, bv.v, (short)0, o[t4], false, false);
    }
  }

  const float sc = 0.04419417382415922f;   // 1/sqrt(512)  (ref: softmax then /sqrt(D))
#pragma unroll
  for (int t4 = 0; t4 < 4; ++t4) {
#pragma unroll
    for (int r = 0; r < 8; ++r) {
      int row = qt * 16 + r + 8 * half;
      int col = h * DHEAD + t4 * 16 + n;
      attn_out[(size_t)(b * S + row) * D + col] = o[t4][r] / lrow[r] * sc;
    }
  }
}

// ---------------- residual + LayerNorm (f32 out + bf16 copy) ----------------
__global__ void add_ln_kernel(const float* __restrict__ seq,
                              const float* __restrict__ attn,
                              const float* __restrict__ gamma,
                              const float* __restrict__ beta,
                              float* __restrict__ outf,
                              unsigned short* __restrict__ outb) {
  const int row = blockIdx.x;
  const int tid = threadIdx.x;   // 256
  __shared__ float red[256];
  size_t base = (size_t)row * D_MODEL;
  float x0 = seq[base + tid] + attn[base + tid];
  float x1 = seq[base + 256 + tid] + attn[base + 256 + tid];

  red[tid] = x0 + x1;
  __syncthreads();
  for (int o = 128; o > 0; o >>= 1) { if (tid < o) red[tid] += red[tid + o]; __syncthreads(); }
  float mu = red[0] * (1.0f / D_MODEL);
  __syncthreads();

  float d0 = x0 - mu, d1 = x1 - mu;
  red[tid] = d0 * d0 + d1 * d1;
  __syncthreads();
  for (int o = 128; o > 0; o >>= 1) { if (tid < o) red[tid] += red[tid + o]; __syncthreads(); }
  float var = red[0] * (1.0f / D_MODEL);
  float rs = rsqrtf(var + 1e-5f);

  float y0 = d0 * rs * gamma[tid] + beta[tid];
  float y1 = d1 * rs * gamma[tid + 256] + beta[tid + 256];
  outf[base + tid] = y0;        outb[base + tid] = f2bf(y0);
  outf[base + 256 + tid] = y1;  outb[base + 256 + tid] = f2bf(y1);
}

extern "C" void kernel_launch(void* const* d_in, const int* in_sizes, int n_in,
                              void* d_out, int out_size, void* d_ws, size_t ws_size,
                              hipStream_t stream) {
  const float* seq   = (const float*)d_in[0];
  const float* Wq    = (const float*)d_in[1];
  const float* bq    = (const float*)d_in[2];
  const float* Wk    = (const float*)d_in[3];
  const float* bk    = (const float*)d_in[4];
  const float* Wv    = (const float*)d_in[5];
  const float* bv    = (const float*)d_in[6];
  const float* W1    = (const float*)d_in[7];
  const float* b1    = (const float*)d_in[8];
  const float* W2    = (const float*)d_in[9];
  const float* b2    = (const float*)d_in[10];
  const float* gamma = (const float*)d_in[11];
  const float* beta  = (const float*)d_in[12];

  char* ws = (char*)d_ws;
  size_t off = 0;
  auto alloc = [&](size_t bytes) { size_t o = off; off += (bytes + 255) & ~(size_t)255; return o; };

  const size_t nTD = (size_t)NTOK * D_MODEL;   // 8192*512
  unsigned short* seqb = (unsigned short*)(ws + alloc(nTD * 2));
  unsigned short* qb   = (unsigned short*)(ws + alloc(nTD * 2));
  unsigned short* kbuf = (unsigned short*)(ws + alloc(nTD * 2));
  unsigned*       vq   = (unsigned*)(ws + alloc(nTD * 2));       // key-pair packed V
  unsigned*       wqp  = (unsigned*)(ws + alloc((D_MODEL / 2) * D_MODEL * 4));
  unsigned*       wkp  = (unsigned*)(ws + alloc((D_MODEL / 2) * D_MODEL * 4));
  unsigned*       wvp  = (unsigned*)(ws + alloc((D_MODEL / 2) * D_MODEL * 4));
  unsigned*       w1p  = (unsigned*)(ws + alloc((D_MODEL / 2) * DFF * 4));
  unsigned*       w2p  = (unsigned*)(ws + alloc((DFF / 2) * D_MODEL * 4));
  float*          attnf= (float*)(ws + alloc(nTD * 4));
  float*          outf = (float*)(ws + alloc(nTD * 4));
  unsigned short* outb = (unsigned short*)(ws + alloc(nTD * 2));
  unsigned short* hb   = (unsigned short*)(ws + alloc((size_t)NTOK * DFF * 2));
  (void)ws_size; (void)in_sizes; (void)n_in; (void)out_size;

  // 1) convert activations + pack weights (fragment-linear) to bf16
  cvt_bf16_kernel<<<(nTD + 255) / 256, 256, 0, stream>>>(seq, seqb, (int)nTD);
  pack_weight_kernel<<<((D_MODEL / 2) * D_MODEL + 255) / 256, 256, 0, stream>>>(Wq, wqp, D_MODEL, D_MODEL);
  pack_weight_kernel<<<((D_MODEL / 2) * D_MODEL + 255) / 256, 256, 0, stream>>>(Wk, wkp, D_MODEL, D_MODEL);
  pack_weight_kernel<<<((D_MODEL / 2) * D_MODEL + 255) / 256, 256, 0, stream>>>(Wv, wvp, D_MODEL, D_MODEL);
  pack_weight_kernel<<<((D_MODEL / 2) * DFF + 255) / 256, 256, 0, stream>>>(W1, w1p, D_MODEL, DFF);
  pack_weight_kernel<<<((DFF / 2) * D_MODEL + 255) / 256, 256, 0, stream>>>(W2, w2p, DFF, D_MODEL);

  // 2) QKV projections; V is emitted directly in attention's B-fragment layout
  dim3 gqkv(D_MODEL / 64, NTOK / 128);
  gemm_wmma_kernel<0><<<gqkv, 128, 0, stream>>>(seqb, wqp, bq, NTOK, D_MODEL, D_MODEL, qb, nullptr, nullptr);
  gemm_wmma_kernel<0><<<gqkv, 128, 0, stream>>>(seqb, wkp, bk, NTOK, D_MODEL, D_MODEL, kbuf, nullptr, nullptr);
  gemm_wmma_kernel<3><<<gqkv, 128, 0, stream>>>(seqb, wvp, bv, NTOK, D_MODEL, D_MODEL, (unsigned short*)vq, nullptr, nullptr);

  // 3) flash attention: one wave per (16 queries, head, batch)
  attn_flash_kernel<<<dim3(SEQLEN / 16, BATCH * NHEADS), 32, 0, stream>>>(qb, kbuf, vq, attnf);

  // 4) residual + LayerNorm
  add_ln_kernel<<<NTOK, 256, 0, stream>>>(seq, attnf, gamma, beta, outf, outb);

  // 5) FFN
  gemm_wmma_kernel<1><<<dim3(DFF / 64, NTOK / 128), 128, 0, stream>>>(outb, w1p, b1, NTOK, DFF, D_MODEL, hb, nullptr, nullptr);
  gemm_wmma_kernel<2><<<dim3(D_MODEL / 64, NTOK / 128), 128, 0, stream>>>(hb, w2p, b2, NTOK, D_MODEL, DFF, nullptr, (float*)d_out, outf);
}
